// EH_NodeModel_MultiHead_86088324481788
// MI455X (gfx1250) — matrix operand order repaired and checked by reading, not compiled, not run on hardware
//
#include <hip/hip_runtime.h>
#include <hip/hip_bf16.h>

// ---------------------------------------------------------------------------
// EH_NodeModel_MultiHead for MI455X (gfx1250, wave32, WMMA)
//   edge phase  : bandwidth/atomic bound -> wave-per-edge scatter kernel
//   node phase  : fused GEMM1 + LayerNorm + SiLU + GEMM2 using
//                 v_wmma_f32_16x16x32_bf16 (f32 accumulate, bf16 operands)
//   LDS weight panels stored TRANSPOSED so both A and B fragments are
//   two contiguous 16B runs per lane -> 2x ds_load_b128, bank-conflict free.
//   Staging uses packed v_cvt_pk_bf16_f32 + b32 LDS stores, branchless
//   except the single padded K-tail step.
// ---------------------------------------------------------------------------

typedef __attribute__((ext_vector_type(16))) __bf16 v16bf;
typedef __attribute__((ext_vector_type(8)))  float  v8f;

#define HID     128
#define TEMB    128
#define NHEAD   4
#define XCOLS   (HID + 2)                 // 130
#define IN_DIM  (2*HID + NHEAD + TEMB)    // 388
#define K1_STEPS 13                       // 13 * 32 = 416 (zero-padded K)
#define K2_STEPS 4
#define ROWS    32                        // nodes per block (node kernel)
#define PA      432                       // bf16 pitch of A tile  (16B-aligned rows)
#define PBT     40                        // bf16 pitch of transposed B panel (80B cols)
#define PA2     136                       // bf16 pitch of A2 tile (272B rows)
#define PH      132                       // f32 pitch of h tile
#define LN_EPS  1e-5f

// ---------------------------------------------------------------------------
__global__ void zero_ws_kernel(float* __restrict__ p, size_t n) {
    size_t i = (size_t)blockIdx.x * blockDim.x + threadIdx.x;
    size_t stride = (size_t)gridDim.x * blockDim.x;
    for (; i < n; i += stride) p[i] = 0.0f;
}

// ---------------------------------------------------------------------------
// One wave32 per edge. Lane l owns channels {l, l+32, l+64, l+96}.
__global__ __launch_bounds__(256)
void edge_kernel(const float* __restrict__ x,
                 const int*   __restrict__ ei,     // (2, E)
                 const float* __restrict__ ea,     // (E, 128)
                 const float* __restrict__ Wp,     // (128, 8)
                 const float* __restrict__ bp,     // (8,)
                 float* __restrict__ m_i,          // (N, 128) accum
                 float* __restrict__ curl,         // (N, 4)   accum
                 int nedges) {
    __shared__ float sW[HID * 2 * NHEAD];
    __shared__ float sB[2 * NHEAD];
    int tid = threadIdx.x;
    for (int i = tid; i < HID * 2 * NHEAD; i += 256) sW[i] = Wp[i];
    if (tid < 2 * NHEAD) sB[tid] = bp[tid];
    __syncthreads();

    int wave = tid >> 5;
    int lane = tid & 31;
    long long e = (long long)blockIdx.x * 8 + wave;
    if (e >= nedges) return;

    int row = ei[e];
    int col = ei[(size_t)nedges + e];

    const float* ear = ea + (size_t)e * HID;
    float ev[4];
    float acc[8];
#pragma unroll
    for (int j = 0; j < 8; ++j) acc[j] = 0.0f;

#pragma unroll
    for (int q = 0; q < 4; ++q) {
        int c = lane + (q << 5);
        float v = ear[c];                       // coalesced 128B per step
        ev[q] = v;
        const float4* w4 = (const float4*)(sW + c * 8);   // 32B-aligned
        float4 wa = w4[0], wb = w4[1];
        acc[0] += v * wa.x; acc[1] += v * wa.y;
        acc[2] += v * wa.z; acc[3] += v * wa.w;
        acc[4] += v * wb.x; acc[5] += v * wb.y;
        acc[6] += v * wb.z; acc[7] += v * wb.w;
    }
    // wave32 all-reduce of the 8 projection partials
#pragma unroll
    for (int off = 16; off > 0; off >>= 1) {
#pragma unroll
        for (int j = 0; j < 8; ++j)
            acc[j] += __shfl_xor(acc[j], off, 32);
    }

    float dx = x[(size_t)col * XCOLS + HID]     - x[(size_t)row * XCOLS + HID];
    float dy = x[(size_t)col * XCOLS + HID + 1] - x[(size_t)row * XCOLS + HID + 1];

    if (lane < NHEAD) {
        float h0 = acc[2 * lane]     + sB[2 * lane];     // Hv[:,h,0]
        float h1 = acc[2 * lane + 1] + sB[2 * lane + 1]; // Hv[:,h,1]
        float C  = h1 * dx - h0 * dy;                    // -dy*h0 + dx*h1
        atomicAdd(&curl[(size_t)col * NHEAD + lane], C);
    }
#pragma unroll
    for (int q = 0; q < 4; ++q)
        atomicAdd(&m_i[(size_t)col * HID + lane + (q << 5)], ev[q]);
}

// ---------------------------------------------------------------------------
__device__ __forceinline__ void st_bf16x2(__bf16* dst, float a, float b) {
    *(__hip_bfloat162*)dst = __float22bfloat162_rn(make_float2(a, b));
}
__device__ __forceinline__ void st_bf16x4(__bf16* dst, float4 v) {
    st_bf16x2(dst,     v.x, v.y);
    st_bf16x2(dst + 2, v.z, v.w);
}

// Stage a 32(k) x 128(c) weight panel TRANSPOSED into sPT[c*PBT + k].
// 2x2 micro-tiles: two coalesced float2 loads, two v_cvt_pk_bf16_f32,
// two b32 LDS stores. GUARD only needed for W1's padded K-tail.
template <bool GUARD>
__device__ __forceinline__ void stage_panelT(const float* __restrict__ W, int s,
                                             __bf16* __restrict__ sPT, int tid,
                                             int rows_valid) {
#pragma unroll
    for (int t = 0; t < 4; ++t) {
        int idx = tid + t * 256;          // 0..1023 : 16 k-pairs x 64 col-pairs
        int rp  = idx >> 6;               // k-pair   0..15
        int cp  = idx & 63;               // col-pair 0..63
        int gr  = (s << 5) + 2 * rp;
        float2 v0 = make_float2(0.f, 0.f), v1 = make_float2(0.f, 0.f);
        if (!GUARD || gr < rows_valid)
            v0 = *(const float2*)(W + (size_t)gr * HID + 2 * cp);
        if (!GUARD || gr + 1 < rows_valid)
            v1 = *(const float2*)(W + (size_t)(gr + 1) * HID + 2 * cp);
        // pack along k: (W[gr][c], W[gr+1][c])
        st_bf16x2(&sPT[(2 * cp)     * PBT + 2 * rp], v0.x, v1.x);
        st_bf16x2(&sPT[(2 * cp + 1) * PBT + 2 * rp], v0.y, v1.y);
    }
}

// 32 nodes per block, 8 waves. Fused: gather node_in -> bf16 GEMM1 (K=416)
// -> LayerNorm -> SiLU -> bf16 GEMM2 (K=128) -> +b2 -> store.
__global__ __launch_bounds__(256)
void node_kernel(const float* __restrict__ x,
                 const float* __restrict__ m_i,
                 const float* __restrict__ curl,
                 const float* __restrict__ u,      // (16, 128)
                 const int*   __restrict__ batch,  // (N,)
                 const float* __restrict__ W1,     // (388, 128)
                 const float* __restrict__ b1,
                 const float* __restrict__ ln_g,
                 const float* __restrict__ ln_b,
                 const float* __restrict__ W2,     // (128, 128)
                 const float* __restrict__ b2,
                 float* __restrict__ out,          // (N, 128)
                 int nnodes) {
    __shared__ __bf16 sA[ROWS * PA];      // node_in tile (reused as A2 tile)
    __shared__ __bf16 sPT[HID * PBT];     // TRANSPOSED weight K-panel (col-major)
    __shared__ float  sH[ROWS * PH];      // GEMM1 output (f32)
    __shared__ float  sMu[ROWS], sRs[ROWS];
    __shared__ float  sRed[ROWS][8][2];
    __shared__ int    sBat[ROWS];

    const int tid  = threadIdx.x;
    const int wave = tid >> 5;
    const int lane = tid & 31;
    const int base = blockIdx.x * ROWS;

    if (tid < ROWS) {
        int node = base + tid;
        sBat[tid] = (node < nnodes) ? batch[node] : 0;
    }
    __syncthreads();

    // ---- gather + convert node_in tile (region-split, vectorized) ----
    // h_E: cols 0..127 from x (row pitch 130 floats -> only 8B-aligned: float2)
    for (int idx = tid; idx < ROWS * 64; idx += 256) {
        int r = idx >> 6, q = idx & 63;
        int node = base + r;
        float2 v = make_float2(0.0f, 0.0f);
        if (node < nnodes) v = *(const float2*)(x + (size_t)node * XCOLS + 2 * q);
        st_bf16x2(&sA[r * PA + 2 * q], v.x, v.y);
    }
    // m_i: cols 128..255 (pitch 128 -> float4)
    for (int idx = tid; idx < ROWS * 32; idx += 256) {
        int r = idx >> 5, q = idx & 31;
        int node = base + r;
        float4 v = make_float4(0.0f, 0.0f, 0.0f, 0.0f);
        if (node < nnodes) v = *(const float4*)(m_i + (size_t)node * HID + 4 * q);
        st_bf16x4(&sA[r * PA + HID + 4 * q], v);
    }
    // curl: cols 256..259
    if (tid < ROWS * NHEAD) {
        int r = tid >> 2, j = tid & 3;
        int node = base + r;
        float v = (node < nnodes) ? curl[(size_t)node * NHEAD + j] : 0.0f;
        sA[r * PA + 2 * HID + j] = (__bf16)v;
    }
    // tau: cols 260..387 from u[batch] (float4, 16B-aligned both sides)
    for (int idx = tid; idx < ROWS * 32; idx += 256) {
        int r = idx >> 5, q = idx & 31;
        float4 v = *(const float4*)(u + (size_t)sBat[r] * TEMB + 4 * q);
        st_bf16x4(&sA[r * PA + 2 * HID + NHEAD + 4 * q], v);
    }
    // zero pad: cols 388..415
    for (int idx = tid; idx < ROWS * 7; idx += 256) {
        int r = idx / 7, q = idx - r * 7;
        st_bf16x4(&sA[r * PA + IN_DIM + 4 * q], make_float4(0.f, 0.f, 0.f, 0.f));
    }

    // ISA 7.12.2 lane mappings (wave32)
    const int mrow  = lane & 15;            // A row / B col / D col
    const int khalf = (lane >> 4) << 3;     // A: +8  K for lanes 16..31
    const int bk0   = (lane >> 4) << 3;     // B: +8  K for lanes 16..31
    const int colB  = wave << 4;            // this wave's 16-col tile

    v8f acc0, acc1;
#pragma unroll
    for (int i = 0; i < 8; ++i) { acc0[i] = 0.0f; acc1[i] = 0.0f; }

    // ---- GEMM1: node_in(32x416) @ W1(416x128), 13 K-steps ----
    for (int s = 0; s < K1_STEPS; ++s) {
        __syncthreads();
        if (s < K1_STEPS - 1) stage_panelT<false>(W1, s, sPT, tid, IN_DIM);
        else                  stage_panelT<true >(W1, s, sPT, tid, IN_DIM);
        __syncthreads();

        v16bf a0, a1, bf;
        const __bf16* ap0 = &sA[mrow * PA + (s << 5) + khalf];
        const __bf16* ap1 = &sA[(16 + mrow) * PA + (s << 5) + khalf];
        const __bf16* bp0 = &sPT[(colB + mrow) * PBT + bk0];
#pragma unroll
        for (int v = 0; v < 8; ++v) {
            int k = ((v & 3) << 1) + ((v >> 2) << 4);   // contiguous 8+8
            a0[2 * v] = ap0[k];  a0[2 * v + 1] = ap0[k + 1];
            a1[2 * v] = ap1[k];  a1[2 * v + 1] = ap1[k + 1];
        }
#pragma unroll
        for (int i = 0; i < 16; ++i) bf[i] = bp0[(i & 7) + ((i >> 3) << 4)];    // 8+8 split, matches A
        acc0 = __builtin_amdgcn_wmma_f32_16x16x32_bf16(false, a0, false, bf,
                                                       (short)0, acc0, false, false);
        acc1 = __builtin_amdgcn_wmma_f32_16x16x32_bf16(false, a1, false, bf,
                                                       (short)0, acc1, false, false);
    }
    __syncthreads();

    // ---- h = acc + b1 -> LDS (f32) ----
    {
        float bias = b1[colB + mrow];
#pragma unroll
        for (int r = 0; r < 8; ++r) {
            int m = r + khalf;                      // D: M = r + 8*(lane>=16)
            sH[m * PH + colB + mrow]        = acc0[r] + bias;
            sH[(16 + m) * PH + colB + mrow] = acc1[r] + bias;
        }
    }
    __syncthreads();

    // ---- LayerNorm stats: 8 threads per row, two-stage reduce ----
    {
        int r = tid >> 3, j = tid & 7;
        float s1 = 0.0f, s2 = 0.0f;
        int c0 = j << 4;
#pragma unroll
        for (int c = 0; c < 16; ++c) {
            float v = sH[r * PH + c0 + c];
            s1 += v; s2 += v * v;
        }
        sRed[r][j][0] = s1; sRed[r][j][1] = s2;
    }
    __syncthreads();
    if (tid < ROWS) {
        float s1 = 0.0f, s2 = 0.0f;
#pragma unroll
        for (int j = 0; j < 8; ++j) { s1 += sRed[tid][j][0]; s2 += sRed[tid][j][1]; }
        float mu  = s1 * (1.0f / HID);
        float var = s2 * (1.0f / HID) - mu * mu;
        sMu[tid] = mu;
        sRs[tid] = rsqrtf(var + LN_EPS);
    }
    __syncthreads();

    // ---- LN + SiLU -> bf16 A2 tile (reuse sA, pitch PA2) ----
    __bf16* sA2 = sA;
    for (int idx = tid; idx < ROWS * 64; idx += 256) {
        int r = idx >> 6, q = idx & 63;         // col-pair q
        int c = 2 * q;
        float va = (sH[r * PH + c]     - sMu[r]) * sRs[r] * ln_g[c]     + ln_b[c];
        float vb = (sH[r * PH + c + 1] - sMu[r]) * sRs[r] * ln_g[c + 1] + ln_b[c + 1];
        va = va / (1.0f + __expf(-va));         // SiLU
        vb = vb / (1.0f + __expf(-vb));
        st_bf16x2(&sA2[r * PA2 + c], va, vb);
    }

    // ---- GEMM2: h(32x128) @ W2(128x128), 4 K-steps ----
    v8f c0v, c1v;
#pragma unroll
    for (int i = 0; i < 8; ++i) { c0v[i] = 0.0f; c1v[i] = 0.0f; }

    for (int s = 0; s < K2_STEPS; ++s) {
        __syncthreads();
        stage_panelT<false>(W2, s, sPT, tid, HID);
        __syncthreads();

        v16bf a0, a1, bf;
        const __bf16* ap0 = &sA2[mrow * PA2 + (s << 5) + khalf];
        const __bf16* ap1 = &sA2[(16 + mrow) * PA2 + (s << 5) + khalf];
        const __bf16* bp0 = &sPT[(colB + mrow) * PBT + bk0];
#pragma unroll
        for (int v = 0; v < 8; ++v) {
            int k = ((v & 3) << 1) + ((v >> 2) << 4);
            a0[2 * v] = ap0[k];  a0[2 * v + 1] = ap0[k + 1];
            a1[2 * v] = ap1[k];  a1[2 * v + 1] = ap1[k + 1];
        }
#pragma unroll
        for (int i = 0; i < 16; ++i) bf[i] = bp0[(i & 7) + ((i >> 3) << 4)];
        c0v = __builtin_amdgcn_wmma_f32_16x16x32_bf16(false, a0, false, bf,
                                                      (short)0, c0v, false, false);
        c1v = __builtin_amdgcn_wmma_f32_16x16x32_bf16(false, a1, false, bf,
                                                      (short)0, c1v, false, false);
    }

    // ---- + b2, store ----
    {
        float bias = b2[colB + mrow];
#pragma unroll
        for (int r = 0; r < 8; ++r) {
            int m  = r + khalf;
            int n0 = base + m;
            int n1 = base + 16 + m;
            if (n0 < nnodes) out[(size_t)n0 * HID + colB + mrow] = c0v[r] + bias;
            if (n1 < nnodes) out[(size_t)n1 * HID + colB + mrow] = c1v[r] + bias;
        }
    }
}

// ---------------------------------------------------------------------------
extern "C" void kernel_launch(void* const* d_in, const int* in_sizes, int n_in,
                              void* d_out, int out_size, void* d_ws, size_t ws_size,
                              hipStream_t stream) {
    const float* x     = (const float*)d_in[0];
    const int*   ei    = (const int*)  d_in[1];
    const float* ea    = (const float*)d_in[2];
    const float* u     = (const float*)d_in[3];
    const int*   batch = (const int*)  d_in[4];
    const float* Wp    = (const float*)d_in[5];
    const float* bp    = (const float*)d_in[6];
    const float* W1    = (const float*)d_in[7];
    const float* b1    = (const float*)d_in[8];
    const float* lng   = (const float*)d_in[9];
    const float* lnb   = (const float*)d_in[10];
    const float* W2    = (const float*)d_in[11];
    const float* b2    = (const float*)d_in[12];
    float* out = (float*)d_out;

    int nnodes = in_sizes[0] / XCOLS;     // 100000
    int nedges = in_sizes[2] / HID;       // 1600000

    float* m_i  = (float*)d_ws;                       // (N,128)
    float* curl = m_i + (size_t)nnodes * HID;         // (N,4)

    size_t zcount = (size_t)nnodes * (HID + NHEAD);
    zero_ws_kernel<<<2048, 256, 0, stream>>>(m_i, zcount);

    edge_kernel<<<(nedges + 7) / 8, 256, 0, stream>>>(
        x, ei, ea, Wp, bp, m_i, curl, nedges);

    node_kernel<<<(nnodes + ROWS - 1) / ROWS, 256, 0, stream>>>(
        x, m_i, curl, u, batch, W1, b1, lng, lnb, W2, b2, out, nnodes);
}
